// HGCLAMIR_66649302499494
// MI455X (gfx1250) — matrix operand order repaired and checked by reading, not compiled, not run on hardware
//
#include <hip/hip_runtime.h>
#include <math.h>

typedef __attribute__((ext_vector_type(16))) __bf16 v16bf;
typedef __attribute__((ext_vector_type(8)))  float  v8f;

#define TAU_F 0.5f
#define ALPHA_F 0.8f

// ---------------- f32 -> bf16 converter ----------------
__global__ void cvt_bf16_kernel(const float* __restrict__ in,
                                __bf16* __restrict__ out, int n) {
    int i = blockIdx.x * blockDim.x + threadIdx.x;
    if (i < n) out[i] = (__bf16)in[i];
}

// ---------------- bf16 WMMA GEMM ----------------
// C[M,N] = act(A[M,K] @ B[K,N] + bias), A/B row-major bf16, accum f32.
// act: 0 = none, 1 = leaky_relu(0.25), 2 = elu
// Cf (f32, stride ldcf) and/or Cb (bf16, stride ldcb) may be null.
// Requires: M % 64 == 0, N % 128 == 0, K % 32 == 0.
__global__ __launch_bounds__(256)
void gemm_bf16_wmma(const __bf16* __restrict__ A, int lda,
                    const __bf16* __restrict__ B, int ldb,
                    const float* __restrict__ bias,
                    float* __restrict__ Cf, int ldcf,
                    __bf16* __restrict__ Cb, int ldcb,
                    int K, int act)
{
    __shared__ __bf16 As[64 * 32];     // [m][k]
    __shared__ __bf16 Bs[128 * 32];    // [n][k]  (K contiguous per output column)

    const int tid  = threadIdx.x;
    const int lane = tid & 31;
    const int wave = tid >> 5;
    const int wm   = wave >> 2;        // 0..1 : wave row (32 rows each)
    const int wn   = wave & 3;         // 0..3 : wave col (32 cols each)
    const int lrow = lane & 15;
    const int hi   = lane >> 4;        // lane group 0/1

    const int tileM = blockIdx.y * 64;
    const int tileN = blockIdx.x * 128;

    v8f acc[2][2] = {};

    // cooperative-load assignments
    const int rowA = tid >> 2;           // 0..63
    const int colA = (tid & 3) * 8;      // 0,8,16,24  (16B chunk per thread)
    const int k2   = (tid >> 4) * 2;     // 0,2,..,30  (k pair per thread)
    const int nB   = (tid & 15) * 8;     // 0..120

    // per-lane LDS byte offset for the async A-tile copy (16B per lane)
    const unsigned ldsA = (unsigned)(uintptr_t)&As[rowA * 32 + colA];
    const __bf16* gA = A + (size_t)(tileM + rowA) * lda + colA;

    for (int k0 = 0; k0 < K; k0 += 32) {
        __syncthreads();

        // ---- A tile 64x32: async global -> LDS, 16B per lane (ASYNCcnt) ----
        {
            unsigned long long ga = (unsigned long long)(uintptr_t)(gA + k0);
            asm volatile("global_load_async_to_lds_b128 %0, %1, off"
                         :: "v"(ldsA), "v"(ga) : "memory");
        }

        // ---- B tile 32x128 -> Bs[n][k] with pair-packed b32 stores ----
        {
            const __bf16* g0 = B + (size_t)(k0 + k2) * ldb + tileN + nB;
            uint4 r0 = *(const uint4*)g0;          // row k2,   n .. n+7
            uint4 r1 = *(const uint4*)(g0 + ldb);  // row k2+1, n .. n+7
            unsigned short u0[8], u1[8];
            *(uint4*)u0 = r0;
            *(uint4*)u1 = r1;
            #pragma unroll
            for (int i = 0; i < 8; ++i) {
                unsigned p = (unsigned)u0[i] | ((unsigned)u1[i] << 16);
                *(unsigned*)&Bs[(nB + i) * 32 + k2] = p;
            }
        }

        // prefetch next K tile while this one is consumed
        if (k0 + 32 < K) {
            __builtin_prefetch(gA + k0 + 32, 0, 1);
            __builtin_prefetch(B + (size_t)(k0 + 32 + k2) * ldb + tileN + nB, 0, 1);
        }

        // async A copy must land before the barrier releases readers
        asm volatile("s_wait_asynccnt 0" ::: "memory");
        __syncthreads();

        // Fragment gathers per ISA 7.12.2 (16-bit A 16x32, B 32x16)
        v16bf afrag[2], bfrag[2];
        #pragma unroll
        for (int s = 0; s < 2; ++s) {
            int r = wm * 32 + s * 16 + lrow;
            #pragma unroll
            for (int e = 0; e < 8; ++e) {
                afrag[s][e]     = As[r * 32 + hi * 8 + e];        // K = hi*8 + e
                afrag[s][8 + e] = As[r * 32 + 16 + hi * 8 + e];   // K = 16 + hi*8 + e
            }
        }
        #pragma unroll
        for (int t = 0; t < 2; ++t) {
            int c = wn * 32 + t * 16 + lrow;
            #pragma unroll
            for (int e = 0; e < 16; ++e)
                bfrag[t][e] = Bs[c * 32 + (e < 8 ? hi * 8 + e : 16 + hi * 8 + (e - 8))];  // K = e<8 ? hi*8 + e : 16 + hi*8 + (e-8)
        }

        #pragma unroll
        for (int s = 0; s < 2; ++s)
            #pragma unroll
            for (int t = 0; t < 2; ++t)
                acc[s][t] = __builtin_amdgcn_wmma_f32_16x16x32_bf16(
                    false, afrag[s], false, bfrag[t],
                    (short)0, acc[s][t], false, false);
    }

    // Epilogue: C layout — VGPR r: lanes0-15 M=r, lanes16-31 M=r+8; N=lane%16
    #pragma unroll
    for (int s = 0; s < 2; ++s) {
        #pragma unroll
        for (int t = 0; t < 2; ++t) {
            int ncol = tileN + wn * 32 + t * 16 + lrow;
            float bv = bias ? bias[ncol] : 0.0f;
            #pragma unroll
            for (int r = 0; r < 8; ++r) {
                int mrow = tileM + wm * 32 + s * 16 + r + hi * 8;
                float v = acc[s][t][r] + bv;
                if (act == 1)      v = (v < 0.0f) ? 0.25f * v : v;
                else if (act == 2) v = (v < 0.0f) ? __expf(v) - 1.0f : v;
                if (Cf) Cf[(size_t)mrow * ldcf + ncol] = v;
                if (Cb) Cb[(size_t)mrow * ldcb + ncol] = (__bf16)v;
            }
        }
    }
}

// ---------------- row L2-normalize (P = 256) ----------------
__global__ __launch_bounds__(256)
void row_normalize(const float* __restrict__ in, float* __restrict__ out) {
    __shared__ float red[256];
    int row = blockIdx.x, t = threadIdx.x;
    float v = in[(size_t)row * 256 + t];
    red[t] = v * v;
    __syncthreads();
    for (int s = 128; s > 0; s >>= 1) {
        if (t < s) red[t] += red[t + s];
        __syncthreads();
    }
    float scale = 1.0f / fmaxf(sqrtf(red[0]), 1e-12f);
    out[(size_t)row * 256 + t] = v * scale;
}

// ---------------- fused InfoNCE row loss ----------------
__global__ __launch_bounds__(256)
void sim_loss(const float* __restrict__ na, const float* __restrict__ nb,
              float weight, float* __restrict__ loss)
{
    __shared__ float va[256];
    __shared__ float red_r[256];
    __shared__ float red_b[256];
    __shared__ float diag[2];
    int i = blockIdx.x, t = threadIdx.x;
    va[t] = na[(size_t)i * 256 + t];
    __syncthreads();
    float sum_r = 0.f, sum_b = 0.f;
    for (int j = t; j < 4096; j += 256) {
        const float* ra = na + (size_t)j * 256;
        const float* rb = nb + (size_t)j * 256;
        float dr = 0.f, db = 0.f;
        for (int k = 0; k < 256; ++k) {
            float x = va[k];
            dr += x * ra[k];
            db += x * rb[k];
        }
        float er = __expf(dr / TAU_F);
        float eb = __expf(db / TAU_F);
        sum_r += er; sum_b += eb;
        if (j == i) { diag[0] = er; diag[1] = eb; }
    }
    red_r[t] = sum_r; red_b[t] = sum_b;
    __syncthreads();
    for (int s = 128; s > 0; s >>= 1) {
        if (t < s) { red_r[t] += red_r[t + s]; red_b[t] += red_b[t + s]; }
        __syncthreads();
    }
    if (t == 0) {
        float l = -logf(diag[1] / (red_r[0] + red_b[0] - diag[0]));
        atomicAdd(loss, weight * l);
    }
}

__global__ void zero_scalar(float* p) { if (threadIdx.x == 0 && blockIdx.x == 0) *p = 0.f; }

// ---------------- host orchestration ----------------
extern "C" void kernel_launch(void* const* d_in, const int* in_sizes, int n_in,
                              void* d_out, int out_size, void* d_ws, size_t ws_size,
                              hipStream_t stream) {
    (void)in_sizes; (void)n_in; (void)out_size; (void)ws_size;
    const int N = 4096, IN = 4096, H = 1024, P = 256;

    const float* x1    = (const float*)d_in[0];
    const float* adj1  = (const float*)d_in[1];
    const float* x2    = (const float*)d_in[2];
    const float* adj2  = (const float*)d_in[3];
    const float* W1a   = (const float*)d_in[4];
    const float* b1a   = (const float*)d_in[5];
    const float* W1b   = (const float*)d_in[6];
    const float* b1b   = (const float*)d_in[7];
    const float* W2a   = (const float*)d_in[8];
    const float* b2a   = (const float*)d_in[9];
    const float* W2b   = (const float*)d_in[10];
    const float* b2b   = (const float*)d_in[11];
    const float* fc1_w = (const float*)d_in[12];
    const float* fc1_b = (const float*)d_in[13];
    const float* fc11_w= (const float*)d_in[14];
    const float* fc11_b= (const float*)d_in[15];
    const float* fc12_w= (const float*)d_in[16];
    const float* fc12_b= (const float*)d_in[17];
    const float* fc2_w = (const float*)d_in[18];
    const float* fc2_b = (const float*)d_in[19];
    const float* fc3_w = (const float*)d_in[20];
    const float* fc3_b = (const float*)d_in[21];

    float* z1   = (float*)d_out;
    float* z2   = z1 + (size_t)N * 2 * H;
    float* loss = z2 + (size_t)N * 2 * H;

    // workspace carve-out
    char* ws = (char*)d_ws;
    size_t off = 0;
    auto carve = [&](size_t bytes) -> void* {
        void* p = ws + off;
        off = (off + bytes + 255) & ~(size_t)255;
        return p;
    };
    __bf16* adjb  = (__bf16*)carve((size_t)N * N * 2);
    __bf16* xb    = (__bf16*)carve((size_t)N * IN * 2);
    __bf16* Wab   = (__bf16*)carve((size_t)IN * H * 2);
    __bf16* Wbb   = (__bf16*)carve((size_t)H * H * 2);
    __bf16* fc1b  = (__bf16*)carve((size_t)2 * H * H * 2);
    __bf16* fc11b = (__bf16*)carve((size_t)H * 2 * P * 2);
    __bf16* fc12b = (__bf16*)carve((size_t)2 * P * P * 2);
    __bf16* fc2b  = (__bf16*)carve((size_t)P * 2 * P * 2);
    __bf16* fc3b  = (__bf16*)carve((size_t)2 * P * P * 2);
    __bf16* t1b   = (__bf16*)carve((size_t)N * H * 2);
    __bf16* zb    = (__bf16*)carve((size_t)N * 2 * H * 2);
    __bf16* p1b   = (__bf16*)carve((size_t)N * H * 2);
    __bf16* p2b   = (__bf16*)carve((size_t)N * 2 * P * 2);
    __bf16* p3b   = (__bf16*)carve((size_t)N * P * 2);
    __bf16* p4b   = (__bf16*)carve((size_t)N * 2 * P * 2);
    float*  hp1   = (float*)carve((size_t)N * P * 4);
    float*  hp2   = (float*)carve((size_t)N * P * 4);
    float*  n1f   = (float*)carve((size_t)N * P * 4);
    float*  n2f   = (float*)carve((size_t)N * P * 4);

    auto cvt = [&](const float* src, __bf16* dst, size_t n) {
        cvt_bf16_kernel<<<(unsigned)((n + 255) / 256), 256, 0, stream>>>(src, dst, (int)n);
    };
    auto gemm = [&](const __bf16* A, int lda, const __bf16* B, int ldb,
                    const float* bias, float* Cf, int ldcf, __bf16* Cb, int ldcb,
                    int M, int Nn, int K, int act) {
        dim3 grid(Nn / 128, M / 64);
        gemm_bf16_wmma<<<grid, 256, 0, stream>>>(A, lda, B, ldb, bias,
                                                 Cf, ldcf, Cb, ldcb, K, act);
    };

    // shared projection weights -> bf16 (once)
    cvt(fc1_w,  fc1b,  (size_t)2 * H * H);
    cvt(fc11_w, fc11b, (size_t)H * 2 * P);
    cvt(fc12_w, fc12b, (size_t)2 * P * P);
    cvt(fc2_w,  fc2b,  (size_t)P * 2 * P);
    cvt(fc3_w,  fc3b,  (size_t)2 * P * P);

    auto view = [&](const float* x, const float* adj,
                    const float* Wa, const float* ba,
                    const float* Wb, const float* bb,
                    float* zout, float* hp) {
        cvt(x,   xb,   (size_t)N * IN);
        cvt(adj, adjb, (size_t)N * N);
        cvt(Wa,  Wab,  (size_t)IN * H);
        cvt(Wb,  Wbb,  (size_t)H * H);
        // t1 = x @ Wa + ba
        gemm(xb, IN, Wab, H, ba, nullptr, 0, t1b, H, N, H, IN, 0);
        // h1 = leaky(G @ t1)  -> z cols [0,H) (f32) + zb cols [0,H) (bf16)
        gemm(adjb, N, t1b, H, nullptr, zout, 2 * H, zb, 2 * H, N, H, N, 1);
        // t2 = h1 @ Wb + bb   (A = zb cols [0,H), lda = 2H)
        gemm(zb, 2 * H, Wbb, H, bb, nullptr, 0, t1b, H, N, H, H, 0);
        // h2 = leaky(G @ t2)  -> z cols [H,2H) + zb cols [H,2H)
        gemm(adjb, N, t1b, H, nullptr, zout + H, 2 * H, zb + H, 2 * H, N, H, N, 1);
        // projection head (ELU chain), final layer no activation -> f32 hp
        gemm(zb, 2 * H, fc1b, H, fc1_b, nullptr, 0, p1b, H, N, H, 2 * H, 2);
        gemm(p1b, H, fc11b, 2 * P, fc11_b, nullptr, 0, p2b, 2 * P, N, 2 * P, H, 2);
        gemm(p2b, 2 * P, fc12b, P, fc12_b, nullptr, 0, p3b, P, N, P, 2 * P, 2);
        gemm(p3b, P, fc2b, 2 * P, fc2_b, nullptr, 0, p4b, 2 * P, N, 2 * P, P, 2);
        gemm(p4b, 2 * P, fc3b, P, fc3_b, hp, P, nullptr, 0, N, P, 2 * P, 0);
    };

    view(x1, adj1, W1a, b1a, W1b, b1b, z1, hp1);
    view(x2, adj2, W2a, b2a, W2b, b2b, z2, hp2);

    row_normalize<<<N, 256, 0, stream>>>(hp1, n1f);
    row_normalize<<<N, 256, 0, stream>>>(hp2, n2f);
    zero_scalar<<<1, 1, 0, stream>>>(loss);
    sim_loss<<<N, 256, 0, stream>>>(n1f, n2f, ALPHA_F, loss);
    sim_loss<<<N, 256, 0, stream>>>(n2f, n1f, 1.0f - ALPHA_F, loss);
}
